// AttnMech_83356725280986
// MI455X (gfx1250) — compile-verified, matrix operand order checked by reading, not run on hardware
//
#include <hip/hip_runtime.h>
#include <hip/hip_bf16.h>

// ---------------- CDNA5 WMMA types ----------------
typedef __attribute__((ext_vector_type(16))) __bf16 v16bf;
typedef __attribute__((ext_vector_type(8)))  float  v8f;

union AFrag { v16bf v; unsigned int u[8]; };

// f32 -> bf16 (RNE) via hardware convert (v_cvt_pk_bf16_f32 on gfx1250)
__device__ __forceinline__ unsigned short f2bf(float f) {
    __bf16 h = (__bf16)f;
    return __builtin_bit_cast(unsigned short, h);
}

__device__ __forceinline__ uint2 pack4(const unsigned short* s) {
    uint2 t;
    t.x = (unsigned)s[0] | ((unsigned)s[1] << 16);
    t.y = (unsigned)s[2] | ((unsigned)s[3] << 16);
    return t;
}

// fragment K offsets for 16-bit WMMA operands (wave32):
// lanes 0-15: VGPR v holds K pair starting at {0,2,4,6,16,18,20,22}[v]
// lanes 16-31: +8
__device__ __forceinline__ int kofs(int v, int half) {
    return ((v & 3) << 1) + ((v >> 2) << 4) + (half << 3);
}

#define LDT 36   // padded LDS row stride (halfwords): 72B rows -> 8B aligned,
                 // 18 dwords -> r*18 mod 64 distinct for 16 rows (conflict free)

// Problem constants
#define CB   4
#define CC   256
#define CH   96
#define CW   96
#define CHW  (CH*CW)       // 9216
#define CN2  2304          // 48*48
#define CKC  1024          // C*2*2
#define QKVSZ (CB*CC*CN2)  // per-projection element count

// ---------------- kernel 1: weights -> bf16 ----------------
__global__ __launch_bounds__(256) void k_convert(
    const float* __restrict__ qw, const float* __restrict__ kw,
    const float* __restrict__ vw, const float* __restrict__ fw,
    unsigned short* __restrict__ Wqkv, unsigned short* __restrict__ fwb)
{
    int idx = blockIdx.x * 256 + threadIdx.x;
    if      (idx < 262144) Wqkv[idx] = f2bf(qw[idx]);
    else if (idx < 524288) Wqkv[idx] = f2bf(kw[idx - 262144]);
    else if (idx < 786432) Wqkv[idx] = f2bf(vw[idx - 524288]);
    else if (idx < 917504) fwb[idx - 786432] = f2bf(fw[idx - 786432]);
}

// ---------------- kernel 2: fused QKV implicit-GEMM conv ----------------
// M=256, K=1024 (c*4+dy*2+dx), N=B*2304. WG tile 128x128, 8 waves 2(M)x4(N),
// each wave 64M x 32N = 8 accumulators. Software-pipelined global->reg->LDS.
__global__ __launch_bounds__(256) void k_proj(
    const unsigned short* __restrict__ Wqkv,
    const float* __restrict__ app, const float* __restrict__ app_pose,
    const float* __restrict__ pose,
    const float* __restrict__ qb, const float* __restrict__ kb,
    const float* __restrict__ vb,
    unsigned short* __restrict__ QKV)
{
    __shared__ unsigned short lA[128 * LDT];
    __shared__ unsigned short lB[128 * LDT];

    const int z = blockIdx.z;
    const float* src  = (z == 0) ? pose : (z == 1) ? app_pose : app;
    const float* bias = (z == 0) ? qb   : (z == 1) ? kb       : vb;
    const unsigned short* Wm = Wqkv + z * (256 * 1024);
    unsigned short* dst = QKV + z * QKVSZ;

    const int m0 = blockIdx.y * 128;
    const int n0 = blockIdx.x * 128;
    const int tid = threadIdx.x;
    const int wave = tid >> 5, lane = tid & 31;
    const int waveM = wave & 1, waveN = wave >> 1;
    const int half = lane >> 4, l15 = lane & 15;

    // staging ownership: thread owns 16 contiguous halfwords of one LDS row
    const int rA  = tid >> 1,   kAh = (tid & 1) * 16;   // A: 128 rows x 2 halves
    const int colB = tid & 127, kBq = (tid >> 7) * 16;  // B: 128 cols x 2 k-quarters

    // B geometry (loop invariant): column -> input patch origin
    const int pB = n0 + colB;
    const int bB = pB / CN2, iiB = pB % CN2;
    const int yB = iiB / 48, xB = iiB % 48;
    const float* srcB = src + ((bB * CC) * CH + yB * 2) * CW + xB * 2;

    uint4 regA[2];
    unsigned short regB[16];

    // prologue: tile k0 = 0
    {
        const uint4* pA = (const uint4*)&Wm[(m0 + rA) * CKC + kAh];
        regA[0] = pA[0]; regA[1] = pA[1];
        const int cb = kBq >> 2;
        #pragma unroll
        for (int i = 0; i < 16; ++i)
            regB[i] = f2bf(srcB[(cb + (i >> 2)) * CHW + ((i >> 1) & 1) * CW + (i & 1)]);
    }

    v8f acc[4][2] = {};

    for (int k0 = 0; k0 < CKC; k0 += 32) {
        __syncthreads();
        {
            uint2 t;
            t.x = regA[0].x; t.y = regA[0].y; *(uint2*)&lA[rA * LDT + kAh]      = t;
            t.x = regA[0].z; t.y = regA[0].w; *(uint2*)&lA[rA * LDT + kAh + 4]  = t;
            t.x = regA[1].x; t.y = regA[1].y; *(uint2*)&lA[rA * LDT + kAh + 8]  = t;
            t.x = regA[1].z; t.y = regA[1].w; *(uint2*)&lA[rA * LDT + kAh + 12] = t;
            *(uint2*)&lB[colB * LDT + kBq]      = pack4(regB + 0);
            *(uint2*)&lB[colB * LDT + kBq + 4]  = pack4(regB + 4);
            *(uint2*)&lB[colB * LDT + kBq + 8]  = pack4(regB + 8);
            *(uint2*)&lB[colB * LDT + kBq + 12] = pack4(regB + 12);
        }
        __syncthreads();

        const int kn = k0 + 32;
        if (kn < CKC) {
            const uint4* pA = (const uint4*)&Wm[(m0 + rA) * CKC + kn + kAh];
            regA[0] = pA[0]; regA[1] = pA[1];
            const int cb = (kn + kBq) >> 2;
            #pragma unroll
            for (int i = 0; i < 16; ++i)
                regB[i] = f2bf(srcB[(cb + (i >> 2)) * CHW + ((i >> 1) & 1) * CW + (i & 1)]);
        }

        AFrag bf0, bf1;
        #pragma unroll
        for (int v = 0; v < 8; ++v) {
            bf0.u[v] = *(const unsigned int*)&lB[(waveN * 32 + l15) * LDT + kofs(v, half)];
            bf1.u[v] = *(const unsigned int*)&lB[(waveN * 32 + 16 + l15) * LDT + kofs(v, half)];
        }
        #pragma unroll
        for (int fm = 0; fm < 4; ++fm) {
            AFrag af;
            #pragma unroll
            for (int v = 0; v < 8; ++v)
                af.u[v] = *(const unsigned int*)&lA[(waveM * 64 + fm * 16 + l15) * LDT + kofs(v, half)];
            acc[fm][0] = __builtin_amdgcn_wmma_f32_16x16x32_bf16(
                false, af.v, false, bf0.v, (short)0, acc[fm][0], false, false);
            acc[fm][1] = __builtin_amdgcn_wmma_f32_16x16x32_bf16(
                false, af.v, false, bf1.v, (short)0, acc[fm][1], false, false);
        }
    }

    // epilogue: +bias, store bf16 as [b][oc][i]
    #pragma unroll
    for (int fn = 0; fn < 2; ++fn) {
        int p = n0 + waveN * 32 + fn * 16 + l15;
        int b = p / CN2, i = p % CN2;
        #pragma unroll
        for (int fm = 0; fm < 4; ++fm) {
            int mb = m0 + waveM * 64 + fm * 16 + half * 8;
            #pragma unroll
            for (int r = 0; r < 8; ++r) {
                int m = mb + r;
                dst[(b * CC + m) * CN2 + i] = f2bf(acc[fm][fn][r] + bias[m]);
            }
        }
    }
}

// ---------------- kernel 3: S = gamma/8 * K @ V^T  (64x64 per (b,h)) ----------------
// one WG (128 thr, 4 waves) per (b,h); tiles staged via async global->LDS DMA.
__global__ __launch_bounds__(128) void k_kv(
    const unsigned short* __restrict__ QKV,
    const float* __restrict__ gamma,
    float* __restrict__ S)
{
    __shared__ unsigned short lK[64 * LDT];
    __shared__ unsigned short lV[64 * LDT];

    const int bh = blockIdx.x;
    const int b = bh >> 2, h = bh & 3;
    const unsigned short* Km = QKV + 1 * QKVSZ + (b * CC + h * 64) * CN2;
    const unsigned short* Vm = QKV + 2 * QKVSZ + (b * CC + h * 64) * CN2;

    const int tid = threadIdx.x;
    const int wave = tid >> 5, lane = tid & 31;
    const int half = lane >> 4, l15 = lane & 15;

    v8f acc[4] = {};

    for (int n0 = 0; n0 < CN2; n0 += 32) {
        __syncthreads();
        // async DMA: 64 rows x 64B, as per-lane 8B chunks (ASYNCcnt tracked)
        #pragma unroll
        for (int i = 0; i < 4; ++i) {
            int q = tid + i * 128;                 // 0..511 chunk id
            int r = q >> 3, kh = (q & 7) * 4;      // row, halfword offset (8B chunks)
            unsigned ldsK = (unsigned)(unsigned long long)(const void*)&lK[r * LDT + kh];
            unsigned ldsV = (unsigned)(unsigned long long)(const void*)&lV[r * LDT + kh];
            unsigned long long gK = (unsigned long long)(const void*)&Km[r * CN2 + n0 + kh];
            unsigned long long gV = (unsigned long long)(const void*)&Vm[r * CN2 + n0 + kh];
            asm volatile("global_load_async_to_lds_b64 %0, %1, off"
                         :: "v"(ldsK), "v"(gK) : "memory");
            asm volatile("global_load_async_to_lds_b64 %0, %1, off"
                         :: "v"(ldsV), "v"(gV) : "memory");
        }
        asm volatile("s_wait_asynccnt 0" ::: "memory");
        __syncthreads();

        AFrag af;
        #pragma unroll
        for (int v = 0; v < 8; ++v)
            af.u[v] = *(const unsigned int*)&lK[(wave * 16 + l15) * LDT + kofs(v, half)];
        #pragma unroll
        for (int jt = 0; jt < 4; ++jt) {
            AFrag bf;
            #pragma unroll
            for (int v = 0; v < 8; ++v)
                bf.u[v] = *(const unsigned int*)&lV[(jt * 16 + l15) * LDT + kofs(v, half)];
            acc[jt] = __builtin_amdgcn_wmma_f32_16x16x32_bf16(
                false, af.v, false, bf.v, (short)0, acc[jt], false, false);
        }
    }

    float sc = gamma[h] * 0.125f;   // 1/sqrt(64) folded in
    #pragma unroll
    for (int jt = 0; jt < 4; ++jt) {
        #pragma unroll
        for (int r = 0; r < 8; ++r) {
            int m = wave * 16 + half * 8 + r;      // c' row
            S[bh * 4096 + m * 64 + jt * 16 + l15] = acc[jt][r] * sc;
        }
    }
}

// ---------------- kernel 4: AttnOut = S^T @ Q   (64 x 2304 per (b,h)) ----------------
__global__ __launch_bounds__(128) void k_attnout(
    const unsigned short* __restrict__ QKV,
    const float* __restrict__ S,
    unsigned short* __restrict__ AO)
{
    __shared__ unsigned short lA[64 * LDT];  // A[m=c][k=c'] = S[c'][c]
    __shared__ unsigned short lB[64 * LDT];  // B[col=i][k=c'] = Q[c'][i]

    const int bh = blockIdx.y;
    const int b = bh >> 2, h = bh & 3;
    const int n0 = blockIdx.x * 64;
    const unsigned short* Qm = QKV + (b * CC + h * 64) * CN2;
    const float* Sm = S + bh * 4096;

    const int tid = threadIdx.x;
    const int wave = tid >> 5, lane = tid & 31;
    const int half = lane >> 4, l15 = lane & 15;

    const int mT = tid & 63, kq = (tid >> 6) * 16;   // thread owns 16 contiguous k of row mT

    v8f acc[4] = {};

    #pragma unroll
    for (int k0 = 0; k0 < 64; k0 += 32) {
        __syncthreads();
        {
            unsigned short ra[16], rb[16];
            #pragma unroll
            for (int i = 0; i < 16; ++i) {
                int kk = k0 + kq + i;
                ra[i] = f2bf(Sm[kk * 64 + mT]);
                rb[i] = Qm[kk * CN2 + n0 + mT];
            }
            #pragma unroll
            for (int j = 0; j < 4; ++j) {
                *(uint2*)&lA[mT * LDT + kq + 4 * j] = pack4(ra + 4 * j);
                *(uint2*)&lB[mT * LDT + kq + 4 * j] = pack4(rb + 4 * j);
            }
        }
        __syncthreads();

        AFrag bf;
        #pragma unroll
        for (int v = 0; v < 8; ++v)
            bf.u[v] = *(const unsigned int*)&lB[(wave * 16 + l15) * LDT + kofs(v, half)];
        #pragma unroll
        for (int fm = 0; fm < 4; ++fm) {
            AFrag af;
            #pragma unroll
            for (int v = 0; v < 8; ++v)
                af.u[v] = *(const unsigned int*)&lA[(fm * 16 + l15) * LDT + kofs(v, half)];
            acc[fm] = __builtin_amdgcn_wmma_f32_16x16x32_bf16(
                false, af.v, false, bf.v, (short)0, acc[fm], false, false);
        }
    }

    int col = n0 + wave * 16 + l15;
    #pragma unroll
    for (int fm = 0; fm < 4; ++fm) {
        #pragma unroll
        for (int r = 0; r < 8; ++r) {
            int m = fm * 16 + half * 8 + r;        // c
            AO[(b * CC + h * 64 + m) * CN2 + col] = f2bf(acc[fm][r]);
        }
    }
}

// ---------------- kernel 5: final fused upsample+concat+1x1 conv ----------------
// out[b,o,y,x] = fb[o] + sum_{c<256} fw[o,c]*pose + sum fw[o,256+c]*AO[b,c,(y/2)*48+x/2]
// GEMM M=256, K=512, N=B*9216; WG tile 128x128, pipelined staging.
__global__ __launch_bounds__(256) void k_final(
    const unsigned short* __restrict__ fwb,
    const float* __restrict__ pose,
    const unsigned short* __restrict__ AO,
    const float* __restrict__ fb,
    float* __restrict__ out)
{
    __shared__ unsigned short lA[128 * LDT];
    __shared__ unsigned short lB[128 * LDT];

    const int m0 = blockIdx.y * 128;
    const int n0 = blockIdx.x * 128;
    const int tid = threadIdx.x;
    const int wave = tid >> 5, lane = tid & 31;
    const int waveM = wave & 1, waveN = wave >> 1;
    const int half = lane >> 4, l15 = lane & 15;

    const int rA  = tid >> 1,   kAh = (tid & 1) * 16;
    const int colB = tid & 127, kBq = (tid >> 7) * 16;

    const int pB = n0 + colB;
    const int bB = pB / CHW, remB = pB % CHW;
    const int yB = remB / 96, xB = remB % 96;
    const float* poseB = pose + ((bB * CC) * CH + yB) * CW + xB;              // + kk*CHW
    const unsigned short* aoB = AO + (bB * CC) * CN2 + (yB >> 1) * 48 + (xB >> 1); // + (kk-256)*CN2

    uint4 regA[2];
    unsigned short regB[16];

    // prologue: k0 = 0 (always pose side since kBq <= 16 < 256)
    {
        const uint4* pA = (const uint4*)&fwb[(m0 + rA) * 512 + kAh];
        regA[0] = pA[0]; regA[1] = pA[1];
        #pragma unroll
        for (int i = 0; i < 16; ++i)
            regB[i] = f2bf(poseB[(kBq + i) * CHW]);
    }

    v8f acc[4][2] = {};

    for (int k0 = 0; k0 < 512; k0 += 32) {
        __syncthreads();
        {
            uint2 t;
            t.x = regA[0].x; t.y = regA[0].y; *(uint2*)&lA[rA * LDT + kAh]      = t;
            t.x = regA[0].z; t.y = regA[0].w; *(uint2*)&lA[rA * LDT + kAh + 4]  = t;
            t.x = regA[1].x; t.y = regA[1].y; *(uint2*)&lA[rA * LDT + kAh + 8]  = t;
            t.x = regA[1].z; t.y = regA[1].w; *(uint2*)&lA[rA * LDT + kAh + 12] = t;
            *(uint2*)&lB[colB * LDT + kBq]      = pack4(regB + 0);
            *(uint2*)&lB[colB * LDT + kBq + 4]  = pack4(regB + 4);
            *(uint2*)&lB[colB * LDT + kBq + 8]  = pack4(regB + 8);
            *(uint2*)&lB[colB * LDT + kBq + 12] = pack4(regB + 12);
        }
        __syncthreads();

        const int kn = k0 + 32;
        if (kn < 512) {
            const uint4* pA = (const uint4*)&fwb[(m0 + rA) * 512 + kn + kAh];
            regA[0] = pA[0]; regA[1] = pA[1];
            if (kn + kBq < 256) {
                #pragma unroll
                for (int i = 0; i < 16; ++i)
                    regB[i] = f2bf(poseB[(kn + kBq + i) * CHW]);
            } else {
                #pragma unroll
                for (int i = 0; i < 16; ++i)
                    regB[i] = aoB[(kn + kBq + i - 256) * CN2];
            }
        }

        AFrag bf0, bf1;
        #pragma unroll
        for (int v = 0; v < 8; ++v) {
            bf0.u[v] = *(const unsigned int*)&lB[(waveN * 32 + l15) * LDT + kofs(v, half)];
            bf1.u[v] = *(const unsigned int*)&lB[(waveN * 32 + 16 + l15) * LDT + kofs(v, half)];
        }
        #pragma unroll
        for (int fm = 0; fm < 4; ++fm) {
            AFrag af;
            #pragma unroll
            for (int v = 0; v < 8; ++v)
                af.u[v] = *(const unsigned int*)&lA[(waveM * 64 + fm * 16 + l15) * LDT + kofs(v, half)];
            acc[fm][0] = __builtin_amdgcn_wmma_f32_16x16x32_bf16(
                false, af.v, false, bf0.v, (short)0, acc[fm][0], false, false);
            acc[fm][1] = __builtin_amdgcn_wmma_f32_16x16x32_bf16(
                false, af.v, false, bf1.v, (short)0, acc[fm][1], false, false);
        }
    }

    #pragma unroll
    for (int fn = 0; fn < 2; ++fn) {
        int p = n0 + waveN * 32 + fn * 16 + l15;
        int b = p / CHW, rem = p % CHW;
        int y = rem / 96, x = rem % 96;
        #pragma unroll
        for (int fm = 0; fm < 4; ++fm) {
            int mb = m0 + waveM * 64 + fm * 16 + half * 8;
            #pragma unroll
            for (int r = 0; r < 8; ++r) {
                int m = mb + r;
                out[(b * CC + m) * CHW + y * CW + x] = acc[fm][fn][r] + fb[m];
            }
        }
    }
}

// ---------------- host launch ----------------
extern "C" void kernel_launch(void* const* d_in, const int* in_sizes, int n_in,
                              void* d_out, int out_size, void* d_ws, size_t ws_size,
                              hipStream_t stream) {
    const float* app      = (const float*)d_in[0];
    const float* app_pose = (const float*)d_in[1];
    const float* pose     = (const float*)d_in[2];
    const float* qw = (const float*)d_in[3];
    const float* qb = (const float*)d_in[4];
    const float* kw = (const float*)d_in[5];
    const float* kb = (const float*)d_in[6];
    const float* vw = (const float*)d_in[7];
    const float* vb = (const float*)d_in[8];
    const float* gamma = (const float*)d_in[9];
    const float* fw = (const float*)d_in[10];
    const float* fb = (const float*)d_in[11];

    char* ws = (char*)d_ws;
    unsigned short* Wqkv = (unsigned short*)(ws);               //  1,572,864 B
    unsigned short* fwb  = (unsigned short*)(ws + 1572864);     //    262,144 B
    unsigned short* QKV  = (unsigned short*)(ws + 1835008);     // 14,155,776 B
    float*          S    = (float*)         (ws + 15990784);    //    262,144 B
    unsigned short* AO   = (unsigned short*)(ws + 16252928);    //  4,718,592 B (total 20 MB)

    k_convert<<<3584, 256, 0, stream>>>(qw, kw, vw, fw, Wqkv, fwb);
    k_proj   <<<dim3(72, 2, 3), 256, 0, stream>>>(Wqkv, app, app_pose, pose,
                                                  qb, kb, vb, QKV);
    k_kv     <<<16, 128, 0, stream>>>(QKV, gamma, S);
    k_attnout<<<dim3(36, 16), 128, 0, stream>>>(QKV, S, AO);
    k_final  <<<dim3(288, 2), 256, 0, stream>>>(fwb, pose, AO, fb, (float*)d_out);
}
